// EarthAttention3DAbsolute_16595753631914
// MI455X (gfx1250) — matrix operand order repaired and checked by reading, not compiled, not run on hardware
//
#include <hip/hip_runtime.h>
#include <hip/hip_bf16.h>

// ---------------------------------------------------------------------------
// Fused EarthAttention3D for MI455X (gfx1250, wave32, WMMA 16x16x32 bf16).
// One workgroup (9 waves = 288 threads) handles one window b (of 960):
//   phase 1: QKV = x @ WqkvT (bf16 WMMA, K=192 -> 6 chunks), Q scaled,
//            Q/K -> LDS row-major, V -> LDS transposed (d-major, m padded 160)
//            (B-fragments software-pipelined across the 36 output tiles)
//   phase 2: per head: S = Q Kt (1 WMMA/tile, K=32), + earth bias + lon mask,
//            register softmax (shfl_xor butterflies), P -> LDS bf16,
//            O = P Vt (K padded 160 -> 5 WMMAs)
//   phase 3: out = O @ WoutT + b_out (pipelined like phase 1, fp32 stores)
// Occupancy is LDS-limited (283,392 B -> 1 WG/WGP), so VGPRs are spent on
// cross-iteration pipelining to keep the XDL pipe fed from L2.
// Workspace: WqkvT bf16 + WoutT bf16 + bias[64][6][144][144] fp32 = 32.1 MB.
// ---------------------------------------------------------------------------

typedef __attribute__((ext_vector_type(16))) __bf16 v16bf;
typedef __attribute__((ext_vector_type(8)))  __bf16 v8bf;
typedef __attribute__((ext_vector_type(8)))  float  v8f;

#define NTOK   144
#define QK_STR 392   // 384 cols (Q:0..191, K:192..383) + pad, 784B/row (bank-skewed)
#define VT_STR 168   // 144 keys + 24 zero pad (PV K-dim padded to 160)
#define P_STR  168
#define O_STR  200   // 192 cols + pad

static __device__ inline v16bf cat8(v8bf a, v8bf b) {
  return __builtin_shufflevector(a, b, 0,1,2,3,4,5,6,7,8,9,10,11,12,13,14,15);
}

// A-fragment (16x32 bf16, ISA 7.12.2): lane holds row (lane&15); element e maps
// to K = (e/8)*16 + (lane/16)*8 + (e%8)  -> two aligned 16B runs.
static __device__ inline v16bf load_afrag(const __bf16* base, int stride,
                                          int lm, int half, int kb) {
  const __bf16* p = base + lm * stride + kb + half * 8;
  v8bf lo = *(const v8bf*)p;
  v8bf hi = *(const v8bf*)(p + 16);
  return cat8(lo, hi);
}

// B-fragment (32x16 bf16) read from B^T stored N-major: lane holds col (lane&15),
// element e maps to K = (e/8)*16 + (lane/16)*8 + (e%8) -> two aligned 16B runs.
static __device__ inline v16bf load_bfrag(const __bf16* base, int stride,
                                          int lm, int half, int kb) {
  const __bf16* p = base + lm * stride + kb + half * 8;
  v8bf lo = *(const v8bf*)p;
  v8bf hi = *(const v8bf*)(p + 16);
  return cat8(lo, hi);
}

static __device__ inline v8bf cvt8(const float* p) {
  v8bf r;
#pragma unroll
  for (int i = 0; i < 8; ++i) r[i] = (__bf16)p[i];
  return r;
}

static __device__ inline v8f wmma_bf16(v16bf a, v16bf b, v8f c) {
  return __builtin_amdgcn_wmma_f32_16x16x32_bf16(false, a, false, b,
                                                 (short)0, c, false, false);
}

// ---------------------------------------------------------------------------
// Prep kernels: bf16 weight transposes + earth-position bias gather.
// ---------------------------------------------------------------------------
__global__ void prep_weights(const float* __restrict__ wqkv,
                             const float* __restrict__ wout,
                             __bf16* __restrict__ wqkvT,
                             __bf16* __restrict__ woutT) {
  int i = blockIdx.x * blockDim.x + threadIdx.x;
  if (i < 576 * 192) {
    int n = i / 192, k = i % 192;
    wqkvT[i] = (__bf16)wqkv[k * 576 + n];          // (576,192) = Wqkv^T
  } else if (i < 576 * 192 + 192 * 192) {
    int j = i - 576 * 192;
    int n = j / 192, k = j % 192;
    woutT[j] = (__bf16)wout[k * 192 + n];          // (192,192) = Wout^T
  }
}

__global__ void prep_bias(const float* __restrict__ table,   // (3312, 64, 6)
                          float* __restrict__ biasM) {       // (64, 6, 144, 144)
  int i = blockIdx.x * blockDim.x + threadIdx.x;
  if (i >= 64 * 6 * NTOK * NTOK) return;
  int m = i % NTOK;
  int n = (i / NTOK) % NTOK;
  int h = (i / (NTOK * NTOK)) % 6;
  int t = i / (NTOK * NTOK * 6);
  int zn = n / 72, hn = (n / 12) % 6, wn = n % 12;
  int zm = m / 72, hm = (m / 12) % 6, wm = m % 12;
  // POS_IDX(n,m) = 828*(zn + 2*zm) + 23*(hn + 6*hm) + (wn - wm + 11)
  int idx = 828 * (zn + 2 * zm) + 23 * (hn + 6 * hm) + (wn - wm + 11);
  biasM[i] = table[(size_t)idx * 384 + t * 6 + h];
}

// ---------------------------------------------------------------------------
// Fused per-window kernel.
// ---------------------------------------------------------------------------
__global__ __launch_bounds__(288)
void earth_attn_fused(const float* __restrict__ x,       // (960,144,192)
                      const float* __restrict__ mask,    // (15,144,144)
                      const float* __restrict__ b_qkv,   // (576)
                      const float* __restrict__ b_out,   // (192)
                      const __bf16* __restrict__ wqkvT,  // (576,192)
                      const __bf16* __restrict__ woutT,  // (192,192)
                      const float* __restrict__ biasM,   // (64,6,144,144)
                      float* __restrict__ out) {         // (960,144,192)
  extern __shared__ char smem[];
  __bf16* qkL = (__bf16*)smem;                 // [144][QK_STR]
  __bf16* vtL = qkL + NTOK * QK_STR;           // [6][32][VT_STR]
  __bf16* pL  = vtL + 6 * 32 * VT_STR;         // [144][P_STR]
  __bf16* oL  = pL + NTOK * P_STR;             // [144][O_STR]

  const int b    = blockIdx.x;
  const int tid  = threadIdx.x;
  const int w    = tid >> 5;        // wave id 0..8 -> 16-row block
  const int l    = tid & 31;
  const int half = l >> 4;
  const int lm   = l & 15;
  const int rows0 = w << 4;

  // ---- phase 0: zero Vt + P regions (zero-padded K tails must read 0) ----
  {
    uint4 z = make_uint4(0u, 0u, 0u, 0u);
    uint4* zp = (uint4*)vtL;   // Vt + P contiguous: 112896 B = 7056 x 16B
    for (int i = tid; i < 7056; i += 288) zp[i] = z;
  }
  __syncthreads();

  // ---- phase 1: QKV projection (double-buffered B-fragments) --------------
  const float* xb = x + (size_t)b * (NTOK * 192);
  v16bf ax[6];
#pragma unroll
  for (int c = 0; c < 6; ++c) {
    const float* p = xb + (rows0 + lm) * 192 + 32 * c + half * 8;
    ax[c] = cat8(cvt8(p), cvt8(p + 16));
  }

  v16bf bmCur[6];
#pragma unroll
  for (int c = 0; c < 6; ++c)
    bmCur[c] = load_bfrag(wqkvT, 192, lm, half, 32 * c);     // tile 0
  float biasCur = b_qkv[lm];

  for (int t = 0; t < 36; ++t) {
    const int colbase = 16 * t;          // uniform across wave
    const int col = colbase + lm;        // per-lane output column (0..575)

    // issue next tile's loads before this tile's WMMA chain (prologue-style)
    const int tn = (t + 1 < 36) ? t + 1 : t;   // clamped, uniform
    v16bf bmNxt[6];
#pragma unroll
    for (int c = 0; c < 6; ++c)
      bmNxt[c] = load_bfrag(wqkvT + 16 * tn * 192, 192, lm, half, 32 * c);
    float biasNxt = b_qkv[16 * tn + lm];

    v8f acc;
#pragma unroll
    for (int r = 0; r < 8; ++r) acc[r] = biasCur;
#pragma unroll
    for (int c = 0; c < 6; ++c) acc = wmma_bf16(ax[c], bmCur[c], acc);

    if (colbase < 192) {                 // Q (pre-scaled by 1/sqrt(hd))
#pragma unroll
      for (int r = 0; r < 8; ++r) {
        int m = rows0 + r + half * 8;
        qkL[m * QK_STR + col] = (__bf16)(acc[r] * 0.17677669529663689f);
      }
    } else if (colbase < 384) {          // K
#pragma unroll
      for (int r = 0; r < 8; ++r) {
        int m = rows0 + r + half * 8;
        qkL[m * QK_STR + col] = (__bf16)acc[r];
      }
    } else {                             // V -> transposed (d-major)
      int hh = (colbase - 384) >> 5;     // uniform
      int dd = (col - 384) & 31;
      __bf16* vrow = vtL + (hh * 32 + dd) * VT_STR;
#pragma unroll
      for (int r = 0; r < 8; ++r) vrow[rows0 + r + half * 8] = (__bf16)acc[r];
    }

#pragma unroll
    for (int c = 0; c < 6; ++c) bmCur[c] = bmNxt[c];
    biasCur = biasNxt;
  }
  __syncthreads();

  // ---- phase 2: attention per head ----------------------------------------
  const int t_idx = b % 64;   // type-of-window (bias reshape: b = a*64 + t)
  const int wi    = b % 15;   // lon window    (mask reshape: b = c*15 + w)
  const float* bmBase = biasM + (size_t)t_idx * 6 * NTOK * NTOK;
  const float* mkBase = mask + (size_t)wi * NTOK * NTOK;

  for (int h = 0; h < 6; ++h) {
    // S = Q K^T : K-dim = head_dim = 32 -> exactly one WMMA per 16x16 tile.
    // All 9 K-fragments loaded (LDS, low latency), 9 independent WMMAs.
    v16bf qa = load_afrag(qkL + rows0 * QK_STR + h * 32, QK_STR, lm, half, 0);
    v16bf kb[9];
#pragma unroll
    for (int j = 0; j < 9; ++j)
      kb[j] = load_bfrag(qkL + 16 * j * QK_STR + 192 + h * 32,
                         QK_STR, lm, half, 0);
    v8f s[9];
#pragma unroll
    for (int j = 0; j < 9; ++j) {
      v8f z;
#pragma unroll
      for (int r = 0; r < 8; ++r) z[r] = 0.0f;
      s[j] = wmma_bf16(qa, kb[j], z);
    }
    // + earth-specific bias + longitude mask
    const float* bmh = bmBase + (size_t)h * NTOK * NTOK;
#pragma unroll
    for (int j = 0; j < 9; ++j) {
#pragma unroll
      for (int r = 0; r < 8; ++r) {
        int nq = rows0 + r + half * 8;
        int m  = 16 * j + lm;
        s[j][r] += bmh[nq * NTOK + m] + mkBase[nq * NTOK + m];
      }
    }
    // register softmax over the 144 keys (row = 16 lanes x 9 regs)
#pragma unroll
    for (int r = 0; r < 8; ++r) {
      float mx = s[0][r];
#pragma unroll
      for (int j = 1; j < 9; ++j) mx = fmaxf(mx, s[j][r]);
      mx = fmaxf(mx, __shfl_xor(mx, 1));
      mx = fmaxf(mx, __shfl_xor(mx, 2));
      mx = fmaxf(mx, __shfl_xor(mx, 4));
      mx = fmaxf(mx, __shfl_xor(mx, 8));
      float sum = 0.0f;
#pragma unroll
      for (int j = 0; j < 9; ++j) {
        float e = __expf(s[j][r] - mx);
        s[j][r] = e;
        sum += e;
      }
      sum += __shfl_xor(sum, 1);
      sum += __shfl_xor(sum, 2);
      sum += __shfl_xor(sum, 4);
      sum += __shfl_xor(sum, 8);
      float inv = 1.0f / sum;
      int nq = rows0 + r + half * 8;
      __bf16* prow = pL + nq * P_STR;
#pragma unroll
      for (int j = 0; j < 9; ++j) prow[16 * j + lm] = (__bf16)(s[j][r] * inv);
    }
    __syncthreads();   // uniform: all waves run all 6 head iterations

    // O = P V^T : K = 144 padded to 160 -> 5 WMMA chunks.
    v16bf pa[5];
#pragma unroll
    for (int c = 0; c < 5; ++c)
      pa[c] = load_afrag(pL + rows0 * P_STR, P_STR, lm, half, 32 * c);
#pragma unroll
    for (int jj = 0; jj < 2; ++jj) {
      v16bf vb[5];
#pragma unroll
      for (int c = 0; c < 5; ++c)
        vb[c] = load_bfrag(vtL + (h * 32 + 16 * jj) * VT_STR,
                           VT_STR, lm, half, 32 * c);
      v8f o;
#pragma unroll
      for (int r = 0; r < 8; ++r) o[r] = 0.0f;
#pragma unroll
      for (int c = 0; c < 5; ++c) o = wmma_bf16(pa[c], vb[c], o);
#pragma unroll
      for (int r = 0; r < 8; ++r) {
        int m = rows0 + r + half * 8;
        oL[m * O_STR + h * 32 + 16 * jj + lm] = (__bf16)o[r];
      }
    }
  }
  __syncthreads();

  // ---- phase 3: output projection (double-buffered like phase 1) ----------
  v16bf ao[6];
#pragma unroll
  for (int c = 0; c < 6; ++c)
    ao[c] = load_afrag(oL + rows0 * O_STR, O_STR, lm, half, 32 * c);
  float* ob = out + (size_t)b * (NTOK * 192);

  v16bf wmCur[6];
#pragma unroll
  for (int c = 0; c < 6; ++c)
    wmCur[c] = load_bfrag(woutT, 192, lm, half, 32 * c);     // tile 0
  float boCur = b_out[lm];

  for (int t = 0; t < 12; ++t) {
    const int tn = (t + 1 < 12) ? t + 1 : t;   // clamped, uniform
    v16bf wmNxt[6];
#pragma unroll
    for (int c = 0; c < 6; ++c)
      wmNxt[c] = load_bfrag(woutT + 16 * tn * 192, 192, lm, half, 32 * c);
    float boNxt = b_out[16 * tn + lm];

    v8f acc;
#pragma unroll
    for (int r = 0; r < 8; ++r) acc[r] = boCur;
#pragma unroll
    for (int c = 0; c < 6; ++c) acc = wmma_bf16(ao[c], wmCur[c], acc);
#pragma unroll
    for (int r = 0; r < 8; ++r) {
      int m = rows0 + r + half * 8;
      ob[m * 192 + 16 * t + lm] = acc[r];
    }

#pragma unroll
    for (int c = 0; c < 6; ++c) wmCur[c] = wmNxt[c];
    boCur = boNxt;
  }
}

// ---------------------------------------------------------------------------
extern "C" void kernel_launch(void* const* d_in, const int* in_sizes, int n_in,
                              void* d_out, int out_size, void* d_ws,
                              size_t ws_size, hipStream_t stream) {
  const float* x         = (const float*)d_in[0];
  const float* mask      = (const float*)d_in[1];
  const float* w_qkv     = (const float*)d_in[2];
  const float* b_qkv     = (const float*)d_in[3];
  const float* w_out     = (const float*)d_in[4];
  const float* b_out     = (const float*)d_in[5];
  const float* bias_tab  = (const float*)d_in[6];

  // workspace layout (32.1 MB total)
  __bf16* wqkvT = (__bf16*)d_ws;                        // 576*192 bf16
  __bf16* woutT = wqkvT + 576 * 192;                    // 192*192 bf16
  float*  biasM = (float*)((char*)d_ws + 294912);       // 64*6*144*144 f32

  prep_weights<<<(576 * 192 + 192 * 192 + 255) / 256, 256, 0, stream>>>(
      w_qkv, w_out, wqkvT, woutT);
  prep_bias<<<(64 * 6 * NTOK * NTOK + 255) / 256, 256, 0, stream>>>(
      bias_tab, biasM);

  const size_t ldsBytes =
      (size_t)(NTOK * QK_STR + 6 * 32 * VT_STR + NTOK * P_STR + NTOK * O_STR) *
      sizeof(__bf16);  // 283,392 B (CDNA5 WGP LDS)
  earth_attn_fused<<<960, 288, ldsBytes, stream>>>(
      x, mask, b_qkv, b_out, wqkvT, woutT, biasM, (float*)d_out);
}
